// MHA_19885698580875
// MI455X (gfx1250) — matrix operand order repaired and checked by reading, not compiled, not run on hardware
//
#include <hip/hip_runtime.h>

#define DEVI __device__ __forceinline__

typedef __bf16 bf16;
typedef __attribute__((ext_vector_type(16))) __bf16 v16bf;
typedef __attribute__((ext_vector_type(8)))  __bf16 v8bf;
typedef __attribute__((ext_vector_type(8)))  float  v8f;

constexpr int Bn = 4, Ns = 2048, Dm = 768, Hh = 12, DK = 64;
constexpr int Mrows = Bn * Ns;  // 8192

// float -> bf16, round-to-nearest-even (bit-level; no reliance on hw cvt codegen)
DEVI bf16 f2bf(float f) {
  unsigned u = __builtin_bit_cast(unsigned, f);
  u += 0x7FFFu + ((u >> 16) & 1u);
  unsigned short h = (unsigned short)(u >> 16);
  return __builtin_bit_cast(bf16, h);
}

// XOR-shuffle within 16-lane groups via ds_swizzle (group-of-32 mode:
// and=0x1F, or=0, xor=MASK; MASK<=8 stays inside a 16-lane half).
template <int MASK> DEVI float swz_xor(float v) {
  int i = __builtin_bit_cast(int, v);
  i = __builtin_amdgcn_ds_swizzle(i, (MASK << 10) | 0x1F);
  return __builtin_bit_cast(float, i);
}
DEVI float redmax16(float v) {
  v = fmaxf(v, swz_xor<1>(v)); v = fmaxf(v, swz_xor<2>(v));
  v = fmaxf(v, swz_xor<4>(v)); v = fmaxf(v, swz_xor<8>(v));
  return v;
}
DEVI float redsum16(float v) {
  v += swz_xor<1>(v); v += swz_xor<2>(v);
  v += swz_xor<4>(v); v += swz_xor<8>(v);
  return v;
}

DEVI v8f wmma_bf16(v16bf a, v16bf b, v8f c) {
  return __builtin_amdgcn_wmma_f32_16x16x32_bf16(
      /*neg_a=*/false, a, /*neg_b=*/false, b,
      /*c_mod=*/(short)0, c, /*reuse_a=*/false, /*reuse_b=*/false);
}

DEVI v16bf cat8(v8bf x, v8bf y) {
  v16bf r;
#pragma unroll
  for (int i = 0; i < 8; i++) { r[i] = x[i]; r[8 + i] = y[i]; }
  return r;
}

// ---------------------------------------------------------------------------
// fp32 -> bf16 elementwise conversion (used for the 4 weight matrices)
// ---------------------------------------------------------------------------
__global__ void cvt_kernel(const float* __restrict__ src, bf16* __restrict__ dst, int n) {
  int i = blockIdx.x * blockDim.x + threadIdx.x;
  int stride = gridDim.x * blockDim.x;
  for (; i < n; i += stride) dst[i] = f2bf(src[i]);
}

// ---------------------------------------------------------------------------
// GEMM: Y[m,n] = sum_k A[m,k] * W[n,k]   (A: [8192,768], W: [768,768] bf16)
// AF32: A is fp32 (converted in-register) else bf16.
// MODE 0: Y -> bf16 [B,H,N,DK]   (q/k heads, row-major over dk)
// MODE 1: Y -> bf16 [B,H,DK,N]   (v heads, transposed for PV B-fragments)
// MODE 2: Y -> fp32 [M,768]      (final output)
// Block = 128 threads (4 waves); each wave owns a 16x64 output strip.
// ---------------------------------------------------------------------------
template <bool AF32, int MODE>
__global__ __launch_bounds__(128) void gemm_kernel(const void* __restrict__ Av,
                                                   const bf16* __restrict__ W,
                                                   void* __restrict__ Outv) {
  const int lane = threadIdx.x & 31;
  const int wave = threadIdx.x >> 5;
  const int l15 = lane & 15;
  const int hi = (lane >> 4) & 1;
  const int m0 = blockIdx.x * 64 + wave * 16;
  const int n0 = blockIdx.y * 64;

  v8f acc[4] = {};
  const int arow = m0 + l15;

  for (int k0 = 0; k0 < Dm; k0 += 32) {
    // A fragment: halves 0-7 -> K = kb..kb+7, halves 8-15 -> K = kb+16..kb+23
    const int kb = k0 + hi * 8;
    v16bf af;
    if constexpr (AF32) {
      const float* A = (const float*)Av;
      const float* p = A + (size_t)arow * Dm + kb;
      v8f f0 = *(const v8f*)p;
      v8f f1 = *(const v8f*)(p + 16);
#pragma unroll
      for (int i = 0; i < 8; i++) { af[i] = f2bf(f0[i]); af[8 + i] = f2bf(f1[i]); }
    } else {
      const bf16* A = (const bf16*)Av;
      const bf16* p = A + (size_t)arow * Dm + kb;
      af = cat8(*(const v8bf*)p, *(const v8bf*)(p + 16));
    }

    // B fragments: lane = N column; halves 0-7 -> K = wk..wk+7, halves 8-15 -> K = wk+16..wk+23
    const int wk = k0 + hi * 8;
#pragma unroll
    for (int nt = 0; nt < 4; nt++) {
      const bf16* wp = W + (size_t)(n0 + nt * 16 + l15) * Dm + wk;
      v16bf bfv = cat8(*(const v8bf*)wp, *(const v8bf*)(wp + 16));
      acc[nt] = wmma_bf16(af, bfv, acc[nt]);
    }
  }

#pragma unroll
  for (int nt = 0; nt < 4; nt++) {
#pragma unroll
    for (int r = 0; r < 8; r++) {
      const int m = m0 + r + hi * 8;
      const int n = n0 + nt * 16 + l15;
      const float v = acc[nt][r];
      if constexpr (MODE == 2) {
        ((float*)Outv)[(size_t)m * Dm + n] = v;
      } else {
        const int b = m >> 11, t = m & 2047;   // m = b*2048 + t
        const int h = n >> 6, dk = n & 63;     // n = h*64 + dk
        bf16* o = (bf16*)Outv;
        if constexpr (MODE == 0)
          o[(((size_t)b * Hh + h) * Ns + t) * DK + dk] = f2bf(v);
        else
          o[(((size_t)b * Hh + h) * DK + dk) * Ns + t] = f2bf(v);
      }
    }
  }
}

// ---------------------------------------------------------------------------
// Flash attention over bf16 heads.
// qh,kh: [B,H,N,DK] bf16 ; vt: [B,H,DK,N] bf16 ; out attn: [B,N,D] bf16.
// Grid: (N/64, B*H); block 128 (4 waves, each owns 16 query rows).
// kv tiles of 32: 4 WMMAs for S (K=DK=64 split in 2), online softmax with
// ds_swizzle row reductions, P relayout via LDS, 4 WMMAs for O (K=32).
// ---------------------------------------------------------------------------
__global__ __launch_bounds__(128) void attn_kernel(const bf16* __restrict__ qh,
                                                   const bf16* __restrict__ kh,
                                                   const bf16* __restrict__ vt,
                                                   bf16* __restrict__ attn) {
  __shared__ alignas(16) bf16 pshare[4][16][32];

  const int lane = threadIdx.x & 31;
  const int wave = threadIdx.x >> 5;
  const int l15 = lane & 15;
  const int hi = (lane >> 4) & 1;
  const int bh = blockIdx.y;
  const int q0 = blockIdx.x * 64 + wave * 16;

  const bf16* qb = qh + (size_t)bh * Ns * DK;
  const bf16* kb = kh + (size_t)bh * Ns * DK;
  const bf16* vb = vt + (size_t)bh * DK * Ns;

  // Q A-fragments for K (=dk) 0..31 and 32..63, loaded once.
  v16bf aq[2];
  {
    const int arow = q0 + l15;
#pragma unroll
    for (int ks = 0; ks < 2; ks++) {
      const bf16* p = qb + (size_t)arow * DK + ks * 32 + hi * 8;
      aq[ks] = cat8(*(const v8bf*)p, *(const v8bf*)(p + 16));
    }
  }

  v8f o[4] = {};
  float mrow[8], lrow[8];
#pragma unroll
  for (int r = 0; r < 8; r++) { mrow[r] = -3.0e30f; lrow[r] = 0.0f; }
  const float scale = 0.125f;  // 1/sqrt(64)

  for (int kv0 = 0; kv0 < Ns; kv0 += 32) {
    // S = Q K^T for 16q x 32kv
    v8f s[2] = {};
#pragma unroll
    for (int st = 0; st < 2; st++) {
      const int krow = kv0 + st * 16 + l15;
#pragma unroll
      for (int ks = 0; ks < 2; ks++) {
        const bf16* p = kb + (size_t)krow * DK + ks * 32 + hi * 8;
        v16bf bfv = cat8(*(const v8bf*)p, *(const v8bf*)(p + 16));
        s[st] = wmma_bf16(aq[ks], bfv, s[st]);
      }
    }

    // online softmax (row stats per accumulator register r)
#pragma unroll
    for (int r = 0; r < 8; r++) {
      float s0 = s[0][r] * scale;
      float s1 = s[1][r] * scale;
      float tm = redmax16(fmaxf(s0, s1));
      float nm = fmaxf(mrow[r], tm);
      float corr = __expf(mrow[r] - nm);
      float p0 = __expf(s0 - nm);
      float p1 = __expf(s1 - nm);
      float ts = redsum16(p0 + p1);
      lrow[r] = lrow[r] * corr + ts;
      mrow[r] = nm;
#pragma unroll
      for (int d = 0; d < 4; d++) o[d][r] *= corr;
      s[0][r] = p0;
      s[1][r] = p1;
    }

    // P: C-layout -> LDS -> A-layout
    __syncthreads();
#pragma unroll
    for (int r = 0; r < 8; r++) {
      pshare[wave][r + hi * 8][l15]      = f2bf(s[0][r]);
      pshare[wave][r + hi * 8][16 + l15] = f2bf(s[1][r]);
    }
    __syncthreads();
    v16bf pf;
    {
      const bf16* p = &pshare[wave][l15][hi * 8];
      pf = cat8(*(const v8bf*)p, *(const v8bf*)(p + 16));
    }

    // O += P * V  (V transposed: lane = dk column reads contiguous kv)
#pragma unroll
    for (int d = 0; d < 4; d++) {
      const bf16* p = vb + (size_t)(d * 16 + l15) * Ns + kv0 + hi * 8;
      v16bf bfv = cat8(*(const v8bf*)p, *(const v8bf*)(p + 16));
      o[d] = wmma_bf16(pf, bfv, o[d]);
    }
  }

  const int b = bh / Hh, h = bh % Hh;
#pragma unroll
  for (int r = 0; r < 8; r++) {
    const float inv = 1.0f / lrow[r];
    const int qrow = q0 + r + hi * 8;
    bf16* op = attn + ((size_t)(b * Ns + qrow)) * Dm + h * DK + l15;
#pragma unroll
    for (int d = 0; d < 4; d++) op[d * 16] = f2bf(o[d][r] * inv);
  }
}

// ---------------------------------------------------------------------------
extern "C" void kernel_launch(void* const* d_in, const int* in_sizes, int n_in,
                              void* d_out, int out_size, void* d_ws, size_t ws_size,
                              hipStream_t stream) {
  const float* Q  = (const float*)d_in[0];
  const float* K  = (const float*)d_in[1];
  const float* V  = (const float*)d_in[2];
  const float* wq = (const float*)d_in[3];
  const float* wk = (const float*)d_in[4];
  const float* wv = (const float*)d_in[5];
  const float* wo = (const float*)d_in[6];
  float* out = (float*)d_out;

  // workspace layout (bf16 elements), all 256B aligned
  bf16* ws = (bf16*)d_ws;
  const size_t Wsz = (size_t)Dm * Dm;        // 589824
  const size_t Psz = (size_t)Bn * Hh * Ns * DK;  // 6291456
  bf16* Wq = ws;
  bf16* Wk = Wq + Wsz;
  bf16* Wv = Wk + Wsz;
  bf16* Wo = Wv + Wsz;
  bf16* qh = Wo + Wsz;
  bf16* khb = qh + Psz;
  bf16* vtb = khb + Psz;
  bf16* at = vtb + Psz;   // attention output [B,N,D] bf16

  cvt_kernel<<<512, 256, 0, stream>>>(wq, Wq, (int)Wsz);
  cvt_kernel<<<512, 256, 0, stream>>>(wk, Wk, (int)Wsz);
  cvt_kernel<<<512, 256, 0, stream>>>(wv, Wv, (int)Wsz);
  cvt_kernel<<<512, 256, 0, stream>>>(wo, Wo, (int)Wsz);

  dim3 gg(Mrows / 64, Dm / 64);  // (128, 12)
  gemm_kernel<true, 0><<<gg, 128, 0, stream>>>(Q, Wq, qh);
  gemm_kernel<true, 0><<<gg, 128, 0, stream>>>(K, Wk, khb);
  gemm_kernel<true, 1><<<gg, 128, 0, stream>>>(V, Wv, vtb);

  dim3 ga(Ns / 64, Bn * Hh);  // (32, 48)
  attn_kernel<<<ga, 128, 0, stream>>>(qh, khb, vtb, at);

  gemm_kernel<false, 2><<<gg, 128, 0, stream>>>(at, Wo, out);
}
